// MultiAttn_68616397521510
// MI455X (gfx1250) — matrix operand order repaired and checked by reading, not compiled, not run on hardware
//
#include <hip/hip_runtime.h>

#define HIDDEN 1024
#define HEADS  16
#define DK     64
#define MAXL   4096
#define BATCH  16

#define L_PER_BLOCK 512   // gridDim.y = MAXL / L_PER_BLOCK = 8
#define WAVES_PER_BLOCK 8 // 256 threads, wave32

typedef __attribute__((ext_vector_type(16))) _Float16 v16h;
typedef __attribute__((ext_vector_type(8)))  _Float16 v8h;
typedef __attribute__((ext_vector_type(8)))  float    v8f;

// ---------------------------------------------------------------------------
// Kernel 0: qpb[b,h,e] = sum_d q[b,h,d] * Wq[h,d,e] + bias[h]
// ---------------------------------------------------------------------------
__global__ void qp_kernel(const float* __restrict__ query,
                          const float* __restrict__ attn,
                          const float* __restrict__ bias,
                          float* __restrict__ qpb)
{
    const int bh = blockIdx.x;
    const int b = bh / HEADS, h = bh % HEADS;
    const int e = threadIdx.x;
    const float* qr = query + (size_t)b * HIDDEN + h * DK;
    const float* wq = attn + (size_t)h * 2 * DK * DK;   // attn[h, :DK, :]
    float acc = 0.0f;
    #pragma unroll 8
    for (int d = 0; d < DK; ++d)
        acc += qr[d] * wq[d * DK + e];
    qpb[bh * DK + e] = acc + bias[h];
}

// ---------------------------------------------------------------------------
// Kernel 1: scores[b,h,l] = sum_e relu( kp[l,e] + qpb[b,h,e] ) * score[h,e]
//
// Transposed mapping: per 16-row L tile compute S^T = Wk^T (64xDK, M=e axis)
// x K^T (DKx16, N=L-row axis) with v_wmma_f32_16x16x32_f16.  The C fragment
// then has e on the VGPR axis and the L-row on the lane axis, so the score
// reduction is in-lane FMAs + ONE shfl_xor(16), and the 16 scores store as
// one coalesced 64B write by lanes 0-15.
// ---------------------------------------------------------------------------
__launch_bounds__(256)
__global__ void scores_kernel(const float* __restrict__ feed,
                              const float* __restrict__ attn,
                              const float* __restrict__ qpb,
                              const float* __restrict__ score,
                              float* __restrict__ scores)
{
    __shared__ _Float16 wkT[DK * DK];   // [e][d]  (transposed, f16)

    const int bh = blockIdx.x;
    const int b = bh / HEADS, h = bh % HEADS;
    const int t = threadIdx.x;

    // Stage Wk[h] = attn[h, DK:, :] into LDS transposed (e-major) + f16.
    const float* wk = attn + (size_t)h * 2 * DK * DK + (size_t)DK * DK;
    for (int i = t; i < DK * DK; i += 256) {
        const int d = i >> 6, e = i & 63;
        wkT[e * DK + d] = (_Float16)wk[i];
    }
    __syncthreads();

    const int lane = t & 31;       // wave32
    const int wave = t >> 5;
    const int lnm  = lane & 15;    // position within half-wave
    const int hi   = lane >> 4;    // half-wave select

    // A-fragments of Wk^T: A is 16x32 f16 (M = e-tile row, K = d).
    // Lane layout: M = lane%16; halves j=0..7 -> K = kbase + hi*16 + j,
    // halves j=8..15 -> K = kbase + hi*16 + j.  Both are contiguous
    // 8-half (16B) runs along d in wkT.
    v16h aW[2][4];   // [kf][mt]
    #pragma unroll
    for (int mt = 0; mt < 4; ++mt) {
        const _Float16* row = &wkT[(mt * 16 + lnm) * DK];
        #pragma unroll
        for (int kf = 0; kf < 2; ++kf) {
            const v8h lo = *(const v8h*)(row + kf * 32 + hi * 16);
            const v8h up = *(const v8h*)(row + kf * 32 + hi * 16 + 8);
            aW[kf][mt] = __builtin_shufflevector(lo, up,
                0, 1, 2, 3, 4, 5, 6, 7, 8, 9, 10, 11, 12, 13, 14, 15);
        }
    }

    // Per-slot epilogue constants: C VGPR j of tile mt holds
    // e = mt*16 + hi*8 + j.  Uniform within each half-wave.
    float qv[4][8], sc[4][8];
    #pragma unroll
    for (int mt = 0; mt < 4; ++mt) {
        #pragma unroll
        for (int j = 0; j < 8; ++j) {
            const int e = mt * 16 + hi * 8 + j;
            qv[mt][j] = qpb[bh * DK + e];
            sc[mt][j] = score[h * DK + e];
        }
    }

    const int iters = L_PER_BLOCK / (16 * WAVES_PER_BLOCK);  // 4
    for (int it = 0; it < iters; ++it) {
        const int l0  = blockIdx.y * L_PER_BLOCK + it * (16 * WAVES_PER_BLOCK)
                      + wave * 16;
        const int row = l0 + lnm;  // this lane's k row (N = lnm)
        const float* rp = feed + ((size_t)row * BATCH + b) * HIDDEN + h * DK;

        // B-fragment (K^T, 32x16 f16): lane n holds N=n, half j -> K =
        // kf*32 + hi*16 + j: a contiguous run of 16 floats per kf.
        float fr[2][16];
        #pragma unroll
        for (int kf = 0; kf < 2; ++kf) {
            #pragma unroll
            for (int q4 = 0; q4 < 4; ++q4) {
                const float4 x = *(const float4*)(rp + kf * 32 + hi * 16 + q4 * 4);
                fr[kf][q4 * 4 + 0] = x.x; fr[kf][q4 * 4 + 1] = x.y;
                fr[kf][q4 * 4 + 2] = x.z; fr[kf][q4 * 4 + 3] = x.w;
            }
        }
        v16h b0, b1;
        #pragma unroll
        for (int j = 0; j < 16; ++j) {
            b0[j] = (_Float16)fr[0][j];
            b1[j] = (_Float16)fr[1][j];
        }

        float p = 0.0f;
        #pragma unroll
        for (int mt = 0; mt < 4; ++mt) {
            v8f c = {};
            c = __builtin_amdgcn_wmma_f32_16x16x32_f16(
                    false, aW[0][mt], false, b0, (short)0, c, false, false);
            c = __builtin_amdgcn_wmma_f32_16x16x32_f16(
                    false, aW[1][mt], false, b1, (short)0, c, false, false);
            // In-lane relu + dot with score over this tile's 8 e-slots.
            #pragma unroll
            for (int j = 0; j < 8; ++j) {
                float en = c[j] + qv[mt][j];
                en = en > 0.0f ? en : 0.0f;
                p += en * sc[mt][j];
            }
        }

        // Combine the two half-wave e-ranges: lane n holds e bits {mt,j},
        // lane n+16 the complementary 8-offset slots.
        p += __shfl_xor(p, 16, 32);

        if (hi == 0) {
            // lanes 0..15 -> rows l0..l0+15: one coalesced 64B store
            scores[(size_t)bh * MAXL + l0 + lnm] = p;
        }
    }
}

// ---------------------------------------------------------------------------
// Kernel 2: per (b,h): softmax over L, then out[d] = sum_l w_l * v[l,b,h,d].
// ---------------------------------------------------------------------------
__launch_bounds__(256)
__global__ void softmax_av_kernel(const float* __restrict__ values,
                                  const float* __restrict__ scores,
                                  float* __restrict__ out)
{
    __shared__ float  w[MAXL];       // 16 KB
    __shared__ float  red[256];
    __shared__ float4 part[256];

    const int bh = blockIdx.x;
    const int b = bh / HEADS, h = bh % HEADS;
    const int t = threadIdx.x;
    const float* s = scores + (size_t)bh * MAXL;

    // Row max
    float loc[16];
    float m = -3.0e38f;
    #pragma unroll
    for (int i = 0; i < 16; ++i) {
        loc[i] = s[t + 256 * i];
        m = fmaxf(m, loc[i]);
    }
    red[t] = m;
    __syncthreads();
    for (int off = 128; off > 0; off >>= 1) {
        if (t < off) red[t] = fmaxf(red[t], red[t + off]);
        __syncthreads();
    }
    m = red[0];
    __syncthreads();

    // exp + sum
    float sum = 0.0f;
    #pragma unroll
    for (int i = 0; i < 16; ++i) {
        const float e = __expf(loc[i] - m);
        w[t + 256 * i] = e;
        sum += e;
    }
    red[t] = sum;
    __syncthreads();
    for (int off = 128; off > 0; off >>= 1) {
        if (t < off) red[t] += red[t + off];
        __syncthreads();
    }
    const float inv = 1.0f / red[0];
    __syncthreads();

    // Weighted sum over V: 16 L-partitions x 16 float4 columns.
    const int p  = t >> 4;          // L partition 0..15
    const int d4 = (t & 15) * 4;    // column group
    float4 acc = make_float4(0.f, 0.f, 0.f, 0.f);
    const float* vb = values + (size_t)b * HIDDEN + h * DK + d4;
    for (int l = p; l < MAXL; l += 16) {
        const float wl = w[l];
        const float4 vv = *(const float4*)(vb + (size_t)l * BATCH * HIDDEN);
        acc.x += wl * vv.x; acc.y += wl * vv.y;
        acc.z += wl * vv.z; acc.w += wl * vv.w;
    }
    part[t] = acc;
    __syncthreads();
    if (t < 16) {
        float4 ssum = make_float4(0.f, 0.f, 0.f, 0.f);
        #pragma unroll
        for (int i = 0; i < 16; ++i) {
            const float4 v2 = part[i * 16 + t];
            ssum.x += v2.x; ssum.y += v2.y; ssum.z += v2.z; ssum.w += v2.w;
        }
        float* op = out + (size_t)b * HIDDEN + h * DK + t * 4;
        *(float4*)op = make_float4(ssum.x * inv, ssum.y * inv,
                                   ssum.z * inv, ssum.w * inv);
    }
}

// ---------------------------------------------------------------------------
extern "C" void kernel_launch(void* const* d_in, const int* in_sizes, int n_in,
                              void* d_out, int out_size, void* d_ws, size_t ws_size,
                              hipStream_t stream)
{
    const float* query  = (const float*)d_in[0];  // (1,B,HIDDEN)
    const float* feed   = (const float*)d_in[1];  // (L,B,HIDDEN)
    const float* values = (const float*)d_in[2];  // (L,B,HIDDEN)
    const float* attn   = (const float*)d_in[3];  // (H,2*DK,DK)
    const float* bias   = (const float*)d_in[4];  // (H,1,1)
    const float* score  = (const float*)d_in[5];  // (H,DK,1)
    float* out = (float*)d_out;

    float* qpb    = (float*)d_ws;                     // B*H*DK floats
    float* scores = qpb + BATCH * HEADS * DK;         // B*H*MAXL floats

    qp_kernel<<<dim3(BATCH * HEADS), dim3(DK), 0, stream>>>(
        query, attn, bias, qpb);

    scores_kernel<<<dim3(BATCH * HEADS, MAXL / L_PER_BLOCK), dim3(256), 0, stream>>>(
        feed, attn, qpb, score, scores);

    softmax_av_kernel<<<dim3(BATCH * HEADS), dim3(256), 0, stream>>>(
        values, scores, out);
}
